// BasicLstm_53480932770696
// MI455X (gfx1250) — matrix operand order repaired and checked by reading, not compiled, not run on hardware
//
#include <hip/hip_runtime.h>

#define BB 64        // batch
#define FF 128       // seq len
#define HH 512       // hidden
#define GG (4*HH)    // gate width
#define NWG 32       // workgroups = HH/16 hidden slices
#define NTH 128      // 4 waves (wave32) = 4 batch tiles of 16
#define KB0 16       // K blocks of 32 over H=512

typedef __attribute__((ext_vector_type(16))) __bf16 v16bf;
typedef __attribute__((ext_vector_type(8)))  float  v8f;

union Frag { uint4 q[2]; v16bf v; };   // one wave32 WMMA A/B fragment: 32B per lane
union H8   { uint4 q; __bf16 h[8]; };

// ---- workspace layout (bytes) ----
static constexpr size_t OFF_SYNC  = 0;                                   // 4 B used
static constexpr size_t OFF_BIAS0 = 256;                                 // GG f32
static constexpr size_t OFF_BIAS1 = OFF_BIAS0 + (size_t)GG*4;
static constexpr size_t OFF_H0    = OFF_BIAS1 + (size_t)GG*4;            // 16640, 256-aligned
static constexpr size_t HBYTES    = (size_t)2*BB*HH*2;                   // ping-pong bf16
static constexpr size_t OFF_H1    = OFF_H0 + HBYTES;
static constexpr size_t OFF_W0    = (OFF_H1 + HBYTES + 1023) & ~(size_t)1023;
static constexpr size_t WBYTES    = (size_t)GG*HH*2;                     // 2 MB each
static constexpr size_t OFF_W1    = OFF_W0 + WBYTES;
static constexpr size_t OFF_W2    = OFF_W1 + WBYTES;                     // total ~6.6 MB

// Zero sync counter + h ping-pong buffers; fuse bias pairs. Re-runs every launch.
__global__ void lstm_init_kernel(const float* __restrict__ b_ih0, const float* __restrict__ b_hh0,
                                 const float* __restrict__ b_ih1, const float* __restrict__ b_hh1,
                                 char* __restrict__ ws) {
    int i = blockIdx.x * blockDim.x + threadIdx.x;
    if (i == 0) *(unsigned*)(ws + OFF_SYNC) = 0u;
    if (i < GG) {
        ((float*)(ws + OFF_BIAS0))[i] = b_ih0[i] + b_hh0[i];
        ((float*)(ws + OFF_BIAS1))[i] = b_ih1[i] + b_hh1[i];
    }
    // zero both h0/h1 ping-pong regions: 2*HBYTES = 262144 B = 65536 dwords
    if (i < 65536) ((unsigned*)(ws + OFF_H0))[i] = 0u;
}

// Pack one (GG x HH) fp32 row-major weight matrix into per-lane bf16 WMMA B-fragments.
// Tile T = ((g*NWG + jt)*KB0 + kb): B[k][n] = W[g*HH + jt*16 + n][kb*32 + k].
// Contiguous k order used here for B (32x16), matching A: lanes 0-15 hold K=0..15 of column n=lane, lanes 16-31 hold K=16..31.
__global__ void lstm_pack_kernel(const float* __restrict__ src, __bf16* __restrict__ dst) {
    int idx = blockIdx.x * blockDim.x + threadIdx.x;      // exactly GG*HH threads
    int e    = idx & 15;                                  // element within lane
    int lane = (idx >> 4) & 31;
    int T    = idx >> 9;                                  // tile index
    int kb   = T & (KB0 - 1);
    int jt   = (T >> 4) & (NWG - 1);
    int g    = T >> 9;
    int row  = g * HH + jt * 16 + (lane & 15);
    int col  = kb * 32 + (lane >> 4) * 16 + e;
    dst[idx] = (__bf16)src[(size_t)row * HH + col];
}

__device__ __forceinline__ float sigm(float x) { return 1.0f / (1.0f + __expf(-x)); }

__device__ __forceinline__ void gsync(unsigned* c, unsigned tgt) {
    __syncthreads();
    __threadfence();                                   // release h stores to device scope
    if (threadIdx.x == 0) {
        atomicAdd(c, 1u);
        while (__hip_atomic_load(c, __ATOMIC_ACQUIRE, __HIP_MEMORY_SCOPE_AGENT) < tgt)
            __builtin_amdgcn_s_sleep(2);
    }
    __syncthreads();                                   // acquire on lead wave invalidated WGP caches
}

// Accumulate 4 gate C-tiles (i,f,g,o) for one (batch-tile, j-slice) over K=512.
// A fragment (16-bit A 16x32): lanes 0-15 hold K 0..15 of row m=lane; lanes 16-31 K 16..31.
__device__ __forceinline__ void mm_acc(const __bf16* __restrict__ hsrc,
                                       const char* __restrict__ wp,
                                       int wg, int lane, int bt,
                                       v8f& a0, v8f& a1, v8f& a2, v8f& a3) {
    const int n16 = lane & 15, hi = lane >> 4;
    const __bf16* hrow = hsrc + (size_t)(bt * 16 + n16) * HH;
    const size_t gs = (size_t)NWG * KB0 * 1024;        // byte stride between gate tile-planes
    const char* tbase = wp + (size_t)(wg * KB0) * 1024 + (size_t)lane * 32;
    for (int kb = 0; kb < KB0; ++kb) {
        Frag a;
        a.q[0] = *(const uint4*)(hrow + kb * 32 + hi * 16);
        a.q[1] = *(const uint4*)(hrow + kb * 32 + hi * 16 + 8);
        const char* tb = tbase + (size_t)kb * 1024;
        Frag b0, b1, b2, b3;
        b0.q[0] = ((const uint4*)(tb         ))[0]; b0.q[1] = ((const uint4*)(tb         ))[1];
        b1.q[0] = ((const uint4*)(tb + gs    ))[0]; b1.q[1] = ((const uint4*)(tb + gs    ))[1];
        b2.q[0] = ((const uint4*)(tb + 2*gs  ))[0]; b2.q[1] = ((const uint4*)(tb + 2*gs  ))[1];
        b3.q[0] = ((const uint4*)(tb + 3*gs  ))[0]; b3.q[1] = ((const uint4*)(tb + 3*gs  ))[1];
        a0 = __builtin_amdgcn_wmma_f32_16x16x32_bf16(false, a.v, false, b0.v, (short)0, a0, false, false);
        a1 = __builtin_amdgcn_wmma_f32_16x16x32_bf16(false, a.v, false, b1.v, (short)0, a1, false, false);
        a2 = __builtin_amdgcn_wmma_f32_16x16x32_bf16(false, a.v, false, b2.v, (short)0, a2, false, false);
        a3 = __builtin_amdgcn_wmma_f32_16x16x32_bf16(false, a.v, false, b3.v, (short)0, a3, false, false);
    }
}

__global__ __launch_bounds__(NTH, 1) void lstm_persistent(
        const float* __restrict__ inputs,   // B,F,4
        const float* __restrict__ W_ih0,    // GG x 4
        const float* __restrict__ W_out,    // 2 x HH
        const float* __restrict__ b_out,    // 2
        char* __restrict__ ws,
        float* __restrict__ out) {          // B,F,2
    __shared__ float xs[BB][4];             // per-WG copy of x_t = [ctx0 ctx1 prev0 prev1]
    const int wg = blockIdx.x, tid = threadIdx.x;
    const int bt = tid >> 5, lane = tid & 31;
    const int n16 = lane & 15, hi = lane >> 4;

    unsigned* syncc = (unsigned*)(ws + OFF_SYNC);
    const float* bias0 = (const float*)(ws + OFF_BIAS0);
    const float* bias1 = (const float*)(ws + OFF_BIAS1);
    __bf16* h0buf = (__bf16*)(ws + OFF_H0);
    __bf16* h1buf = (__bf16*)(ws + OFF_H1);
    const char* w0 = ws + OFF_W0;   // packed W_hh0
    const char* w1 = ws + OFF_W1;   // packed W_ih1
    const char* w2 = ws + OFF_W2;   // packed W_hh1

    // per-lane loop invariants: fused biases and the K=4 W_ih0 rows for this lane's gate column
    float4 wih[4]; float bs0[4], bs1[4];
    #pragma unroll
    for (int g = 0; g < 4; ++g) {
        int j = g * HH + wg * 16 + n16;
        bs0[g] = bias0[j]; bs1[g] = bias1[j];
        wih[g] = ((const float4*)W_ih0)[j];
    }

    // t=0 uses the full 4-channel input row
    for (int i = tid; i < BB * 4; i += NTH)
        xs[i >> 2][i & 3] = inputs[(size_t)((i >> 2) * FF) * 4 + (i & 3)];
    __syncthreads();

    v8f c0 = {}, c1 = {};   // cell state fragments live in VGPRs across all 128 steps

    for (int t = 0; t < FF; ++t) {
        const __bf16* h0p = h0buf + (size_t)(t & 1) * BB * HH;
        __bf16*       h0n = h0buf + (size_t)((t + 1) & 1) * BB * HH;
        const __bf16* h1p = h1buf + (size_t)(t & 1) * BB * HH;
        __bf16*       h1n = h1buf + (size_t)((t + 1) & 1) * BB * HH;

        // ---------- cell 0: gates = h0 @ W_hh0^T + x @ W_ih0^T + bias ----------
        v8f g0 = {}, g1 = {}, g2 = {}, g3 = {};
        mm_acc(h0p, w0, wg, lane, bt, g0, g1, g2, g3);
        #pragma unroll
        for (int r = 0; r < 8; ++r) {
            int b = bt * 16 + hi * 8 + r;                     // C layout: M = r + 8*(lane>=16)
            float x0 = xs[b][0], x1 = xs[b][1], x2 = xs[b][2], x3 = xs[b][3];
            float vi = g0[r] + bs0[0] + x0*wih[0].x + x1*wih[0].y + x2*wih[0].z + x3*wih[0].w;
            float vf = g1[r] + bs0[1] + x0*wih[1].x + x1*wih[1].y + x2*wih[1].z + x3*wih[1].w;
            float vg = g2[r] + bs0[2] + x0*wih[2].x + x1*wih[2].y + x2*wih[2].z + x3*wih[2].w;
            float vo = g3[r] + bs0[3] + x0*wih[3].x + x1*wih[3].y + x2*wih[3].z + x3*wih[3].w;
            float cn = sigm(vf) * c0[r] + sigm(vi) * tanhf(vg);
            c0[r] = cn;
            h0n[(size_t)b * HH + wg * 16 + n16] = (__bf16)(sigm(vo) * tanhf(cn));
        }
        gsync(syncc, (unsigned)(t * 2 + 1) * NWG);

        // ---------- cell 1: gates = h0_new @ W_ih1^T + h1 @ W_hh1^T + bias ----------
        v8f d0 = {}, d1 = {}, d2 = {}, d3 = {};
        mm_acc(h0n, w1, wg, lane, bt, d0, d1, d2, d3);
        mm_acc(h1p, w2, wg, lane, bt, d0, d1, d2, d3);
        #pragma unroll
        for (int r = 0; r < 8; ++r) {
            int b = bt * 16 + hi * 8 + r;
            float vi = d0[r] + bs1[0], vf = d1[r] + bs1[1];
            float vg = d2[r] + bs1[2], vo = d3[r] + bs1[3];
            float cn = sigm(vf) * c1[r] + sigm(vi) * tanhf(vg);
            c1[r] = cn;
            h1n[(size_t)b * HH + wg * 16 + n16] = (__bf16)(sigm(vo) * tanhf(cn));
        }
        gsync(syncc, (unsigned)(t * 2 + 2) * NWG);

        // ---------- output projection, done redundantly per WG (removes a 3rd grid sync) ----------
        {
            int b = tid >> 1, o = tid & 1;                    // 128 threads = 64x2 outputs
            const __bf16* hr = h1n + (size_t)b * HH;
            const float* wo = W_out + o * HH;
            float acc = 0.f;
            for (int j = 0; j < HH; j += 8) {
                H8 v; v.q = *(const uint4*)(hr + j);
                float4 wA = *(const float4*)(wo + j);
                float4 wB = *(const float4*)(wo + j + 4);
                acc += (float)v.h[0]*wA.x + (float)v.h[1]*wA.y + (float)v.h[2]*wA.z + (float)v.h[3]*wA.w
                     + (float)v.h[4]*wB.x + (float)v.h[5]*wB.y + (float)v.h[6]*wB.z + (float)v.h[7]*wB.w;
            }
            acc += b_out[o];
            if (wg == 0) out[((size_t)b * FF + t) * 2 + o] = acc;
            xs[b][2 + o] = acc;                               // feedback for next step
            if (o == 0 && t + 1 < FF) {
                xs[b][0] = inputs[((size_t)b * FF + t + 1) * 4 + 0];
                xs[b][1] = inputs[((size_t)b * FF + t + 1) * 4 + 1];
            }
        }
        __syncthreads();
    }
}

extern "C" void kernel_launch(void* const* d_in, const int* in_sizes, int n_in,
                              void* d_out, int out_size, void* d_ws, size_t ws_size,
                              hipStream_t stream) {
    const float* inputs = (const float*)d_in[0];
    const float* W_ih0  = (const float*)d_in[1];
    const float* W_hh0  = (const float*)d_in[2];
    const float* b_ih0  = (const float*)d_in[3];
    const float* b_hh0  = (const float*)d_in[4];
    const float* W_ih1  = (const float*)d_in[5];
    const float* W_hh1  = (const float*)d_in[6];
    const float* b_ih1  = (const float*)d_in[7];
    const float* b_hh1  = (const float*)d_in[8];
    const float* W_out  = (const float*)d_in[9];
    const float* b_out  = (const float*)d_in[10];
    char* ws = (char*)d_ws;

    hipLaunchKernelGGL(lstm_init_kernel, dim3(256), dim3(256), 0, stream,
                       b_ih0, b_hh0, b_ih1, b_hh1, ws);
    hipLaunchKernelGGL(lstm_pack_kernel, dim3(4096), dim3(256), 0, stream, W_hh0, (__bf16*)(ws + OFF_W0));
    hipLaunchKernelGGL(lstm_pack_kernel, dim3(4096), dim3(256), 0, stream, W_ih1, (__bf16*)(ws + OFF_W1));
    hipLaunchKernelGGL(lstm_pack_kernel, dim3(4096), dim3(256), 0, stream, W_hh1, (__bf16*)(ws + OFF_W2));
    hipLaunchKernelGGL(lstm_persistent, dim3(NWG), dim3(NTH), 0, stream,
                       inputs, W_ih0, W_out, b_out, ws, (float*)d_out);
}
